// TaskAdaptiveLoRALayer_58454504899003
// MI455X (gfx1250) — matrix operand order repaired and checked by reading, not compiled, not run on hardware
//
#include <hip/hip_runtime.h>

typedef __bf16 bf16;
typedef __attribute__((ext_vector_type(16))) __bf16 v16bf;
typedef __attribute__((ext_vector_type(8)))  __bf16 v8bf;
typedef __attribute__((ext_vector_type(8)))  float  v8f;
typedef __attribute__((ext_vector_type(4)))  int    v4i;
typedef __attribute__((ext_vector_type(4)))  unsigned int u32x4;
typedef __attribute__((ext_vector_type(8)))  int    i32x8;
typedef __attribute__((ext_vector_type(4)))  int    i32x4;

typedef __attribute__((address_space(1))) v4i gv4i;   // global <4 x i32>
typedef __attribute__((address_space(3))) v4i lv4i;   // LDS    <4 x i32>

#define B_    4
#define S_    2048
#define DIN   4096
#define DOUT  4096
#define R_    16
#define E_    4
#define T_    32
#define M_TOT (B_ * S_)
#define SCALING 2.0f

// ---- workspace layout (bytes) ----
#define WS_XB     0ull                                   // bf16 x       [M_TOT, DIN]
#define WS_WB     (WS_XB     + (size_t)M_TOT * DIN * 2)  // bf16 W       [DOUT, DIN]
#define WS_LB     (WS_WB     + (size_t)DOUT  * DIN * 2)  // bf16 lora_B  [DOUT, R]
#define WS_PB     (WS_LB     + (size_t)DOUT  * R_  * 2)  // bf16 proj    [M_TOT, R] (pre-scaled)
#define WS_ACOMB  (WS_PB     + (size_t)M_TOT * R_  * 2)  // f32  A_comb  [B, R, DIN]
#define WS_POOLED (WS_ACOMB  + (size_t)B_ * R_ * DIN * 4)// f32  pooled  [B, DIN]
#define WS_WTS    (WS_POOLED + (size_t)B_ * DIN * 4)     // f32  wts     [B, E]

#if __has_builtin(__builtin_amdgcn_tensor_load_to_lds)
#define USE_TDM 1
#else
#define USE_TDM 0
#endif

// ---- CDNA5 async global->LDS copy (ASYNCcnt-tracked) ----
__device__ __forceinline__ void async_copy_b128(void* lds, const void* g) {
#if __has_builtin(__builtin_amdgcn_global_load_async_to_lds_b128)
    __builtin_amdgcn_global_load_async_to_lds_b128((gv4i*)g, (lv4i*)lds, 0, 0);
#else
    asm volatile("global_load_async_to_lds_b128 %0, %1, off"
                 :
                 : "v"((lv4i*)lds), "v"((gv4i*)g)
                 : "memory");
#endif
}

__device__ __forceinline__ void wait_async0() {
#if __has_builtin(__builtin_amdgcn_s_wait_asynccnt)
    __builtin_amdgcn_s_wait_asynccnt(0);
#else
    asm volatile("s_wait_asynccnt 0x0" ::: "memory");
#endif
}

__device__ __forceinline__ void wait_tensor0() {
#if __has_builtin(__builtin_amdgcn_s_wait_tensorcnt)
    __builtin_amdgcn_s_wait_tensorcnt((short)0);
#else
    asm volatile("s_wait_tensorcnt 0x0" ::: "memory");
#endif
}

__device__ __forceinline__ unsigned lds_byte_off(const void* p) {
    return (unsigned)(size_t)(__attribute__((address_space(3))) const void*)p;
}

#if USE_TDM
// One TDM descriptor moves a 128-row x 64B 2D tile (bf16, row stride DIN elems)
// into LDS with 16B padding appended after every 64B row (pad_interval=3 -> 64B,
// pad_amount=3 -> 4 DWORDs), producing the LDT=40 bf16 LDS pitch directly.
__device__ __forceinline__ void tdm_load_tile(unsigned lds_addr, const bf16* gptr) {
    const unsigned long long ga = (unsigned long long)(size_t)gptr;
    // D# group 0: count=1 | lds_addr | global_addr[56:0] | type=2
    u32x4 g0 = { 1u,
                 lds_addr,
                 (unsigned)ga,
                 (unsigned)((ga >> 32) & 0x01ffffffu) | (2u << 30) };
    // D# group 1:
    //  w0: data_size=1(2B)<<16 | pad_enable<<20 | pad_interval=3<<22 | pad_amount=3<<25
    //  w1[31:16] = tensor_dim0 low16 (4096)
    //  w2[31:16] = tensor_dim1 low16 (8192)
    //  w3[31:16] = tile_dim0 (32 elems)
    //  w4[15:0]  = tile_dim1 (128 rows)
    //  w5        = tensor_dim0_stride low32 (4096 elems)
    i32x8 g1 = { (int)((1u << 16) | (1u << 20) | (3u << 22) | (3u << 25)),
                 (int)((unsigned)DIN << 16),
                 (int)((unsigned)M_TOT << 16),
                 (int)(32u << 16),
                 (int)128,
                 (int)DIN,
                 0, 0 };
    i32x4 z4 = {0, 0, 0, 0};
#if __clang_major__ >= 23
    i32x8 z8 = {0, 0, 0, 0, 0, 0, 0, 0};
    __builtin_amdgcn_tensor_load_to_lds(g0, g1, z4, z4, z8, 0);
#else
    __builtin_amdgcn_tensor_load_to_lds(g0, g1, z4, z4, 0);
#endif
}
#endif

// ------------------------------------------------------------------
__global__ void zero_pooled_kernel(float* __restrict__ pooled) {
    pooled[blockIdx.x * 256 + threadIdx.x] = 0.0f;       // grid == B*DIN/256
}

// generic fp32 -> bf16 conversion, 8 elements per thread
__global__ void convert_bf16_kernel(const float* __restrict__ src, bf16* __restrict__ dst) {
    const size_t i = ((size_t)blockIdx.x * 256 + threadIdx.x) * 8;
    float4 f0 = *(const float4*)(src + i);
    float4 f1 = *(const float4*)(src + i + 4);
    v8bf o;
    o[0] = (bf16)f0.x; o[1] = (bf16)f0.y; o[2] = (bf16)f0.z; o[3] = (bf16)f0.w;
    o[4] = (bf16)f1.x; o[5] = (bf16)f1.y; o[6] = (bf16)f1.z; o[7] = (bf16)f1.w;
    *(v8bf*)(dst + i) = o;
}

// convert x -> bf16 and accumulate sum over S for the gate pooling
__global__ void convert_x_pool_kernel(const float* __restrict__ x,
                                      bf16* __restrict__ xb,
                                      float* __restrict__ pooled) {
    const int d  = blockIdx.x * 256 + threadIdx.x;
    const int b  = blockIdx.z;
    const int s0 = blockIdx.y * 128;
    size_t idx = ((size_t)b * S_ + s0) * DIN + d;
    float acc = 0.0f;
    #pragma unroll 4
    for (int s = 0; s < 128; ++s) {
        float v = x[idx];
        xb[idx] = (bf16)v;
        acc += v;
        idx += DIN;
    }
    atomicAdd(&pooled[b * DIN + d], acc);
}

// gate MLP + softmax -> wts[B,E]
__global__ void gate_kernel(const float* __restrict__ pooled,
                            const float* __restrict__ gw1, const float* __restrict__ gb1,
                            const float* __restrict__ gw2, const float* __restrict__ gb2,
                            float* __restrict__ wts) {
    __shared__ float hid[B_ * T_];
    __shared__ float logit[B_ * E_];
    const int t = threadIdx.x;
    if (t < B_ * T_) {
        const int b = t / T_, tt = t % T_;
        const float* p = pooled + b * DIN;
        const float* w = gw1 + tt * DIN;
        float acc = 0.0f;
        for (int d = 0; d < DIN; ++d) acc += p[d] * w[d];
        float h = acc * (1.0f / S_) + gb1[tt];
        hid[t] = h > 0.0f ? h : 0.0f;
    }
    __syncthreads();
    if (t < B_ * E_) {
        const int b = t / E_, e = t % E_;
        float acc = gb2[e];
        for (int k = 0; k < T_; ++k) acc += hid[b * T_ + k] * gw2[e * T_ + k];
        logit[t] = acc;
    }
    __syncthreads();
    if (t < B_) {
        float m = logit[t * E_];
        for (int e = 1; e < E_; ++e) m = fmaxf(m, logit[t * E_ + e]);
        float ex[E_]; float s = 0.0f;
        for (int e = 0; e < E_; ++e) { ex[e] = __expf(logit[t * E_ + e] - m); s += ex[e]; }
        for (int e = 0; e < E_; ++e) wts[t * E_ + e] = ex[e] / s;
    }
}

// A_comb[b,r,d] = A_shared[r,d] + sum_e wts[b,e] * A_experts[e,r,d]
__global__ void acomb_kernel(const float* __restrict__ Ash, const float* __restrict__ Aex,
                             const float* __restrict__ wts, float* __restrict__ acomb) {
    const size_t i = (size_t)blockIdx.x * 256 + threadIdx.x;   // over B*R*DIN
    const int d  = (int)(i % DIN);
    const int br = (int)(i / DIN);
    const int r  = br % R_;
    const int b  = br / R_;
    float v = Ash[r * DIN + d];
    #pragma unroll
    for (int e = 0; e < E_; ++e)
        v += wts[b * E_ + e] * Aex[((size_t)e * R_ + r) * DIN + d];
    acomb[i] = v;
}

// pb[m,r] = SCALING * sum_d xb[m,d] * acomb[b,r,d]   (bf16 out, pre-scaled)
__global__ void proj_kernel(const bf16* __restrict__ xb, const float* __restrict__ acomb,
                            bf16* __restrict__ pb) {
    __shared__ float red[R_][17];
    const int m    = blockIdx.x;
    const int b    = m / S_;
    const int lane = threadIdx.x % 16;
    const int r    = threadIdx.x / 16;
    const bf16*  xr = xb + (size_t)m * DIN;
    const float* ar = acomb + ((size_t)b * R_ + r) * DIN;
    float acc = 0.0f;
    #pragma unroll 8
    for (int d = lane; d < DIN; d += 16)
        acc += (float)xr[d] * ar[d];
    red[r][lane] = acc;
    __syncthreads();
    if (threadIdx.x < R_) {
        float s = 0.0f;
        #pragma unroll
        for (int k = 0; k < 16; ++k) s += red[threadIdx.x][k];
        pb[(size_t)m * R_ + threadIdx.x] = (bf16)(SCALING * s);
    }
}

// ------------------------------------------------------------------
// Main GEMM: out = xb @ wb^T + bias + pb @ lb^T  (bf16 WMMA, fused LoRA,
// TDM (tensor_load_to_lds) double-buffered tile pipeline; async-load fallback)
#define BM 128
#define BN 128
#define BK 32
#define LDT 40                 // padded LDS stride (bf16 units): 80B rows, conflict-free rotation
#define KTILES (DIN / BK)

union AF { v16bf v; v8bf h[2]; };

__global__ __launch_bounds__(256) void gemm_kernel(const bf16* __restrict__ xb,
                                                   const bf16* __restrict__ wb,
                                                   const float* __restrict__ bias,
                                                   const bf16* __restrict__ pb,
                                                   const bf16* __restrict__ lb,
                                                   float* __restrict__ out) {
    __shared__ bf16 As[2][BM * LDT];
    __shared__ bf16 Bs[2][BN * LDT];

    const int tid  = threadIdx.x;
    const int wave = tid / 32, lane = tid % 32;
    const int g    = lane / 16, ln = lane % 16;   // lane-group (K-half) / position
    const int wm   = wave % 4,  wn = wave / 4;    // 4x2 wave grid -> wave tile 32x64
    const int mBlk = blockIdx.y * BM;
    const int nBlk = blockIdx.x * BN;

    v8f acc[2][4];
    #pragma unroll
    for (int i = 0; i < 2; ++i)
        #pragma unroll
        for (int j = 0; j < 4; ++j)
            acc[i][j] = (v8f){0.f, 0.f, 0.f, 0.f, 0.f, 0.f, 0.f, 0.f};

#if USE_TDM
    // Wave 0 drives the Tensor Data Mover: one descriptor per matrix per k-tile.
    auto loadTile = [&](int buf, int kt) {
        if (wave == 0) {
            tdm_load_tile(lds_byte_off(&As[buf][0]), xb + (size_t)mBlk * DIN + kt * BK);
            tdm_load_tile(lds_byte_off(&Bs[buf][0]), wb + (size_t)nBlk * DIN + kt * BK);
        }
    };
    auto waitTiles = [&]() { if (wave == 0) wait_tensor0(); };
#else
    // Per-lane ASYNCcnt path: 64 rows x 32 cols per instruction, 2 per matrix.
    const int lrow = tid / 4;
    const int lcol = (tid % 4) * 8;
    auto loadTile = [&](int buf, int kt) {
        #pragma unroll
        for (int p = 0; p < 2; ++p) {
            const int row = lrow + 64 * p;
            async_copy_b128(&As[buf][row * LDT + lcol],
                            xb + (size_t)(mBlk + row) * DIN + kt * BK + lcol);
            async_copy_b128(&Bs[buf][row * LDT + lcol],
                            wb + (size_t)(nBlk + row) * DIN + kt * BK + lcol);
        }
    };
    auto waitTiles = [&]() { wait_async0(); };
#endif

    loadTile(0, 0);
    waitTiles();
    __syncthreads();

    for (int kt = 0; kt < KTILES; ++kt) {
        const int cur = kt & 1;
        if (kt + 1 < KTILES) loadTile(cur ^ 1, kt + 1);  // DMA overlaps compute below

        const bf16* a0 = &As[cur][0];
        const bf16* b0 = &Bs[cur][0];

        AF a[2], bq[4];
        // A frags: lane g holds K={8g..8g+7} (h[0]) and K={16+8g..23+8g} (h[1])
        #pragma unroll
        for (int fm = 0; fm < 2; ++fm) {
            const bf16* p = a0 + (wm * 32 + fm * 16 + ln) * LDT;
            a[fm].h[0] = *(const v8bf*)(p + 8 * g);
            a[fm].h[1] = *(const v8bf*)(p + 16 + 8 * g);
        }
        // B frags: lane g holds K={8g..8g+7} (h[0]) and K={16+8g..23+8g} (h[1]) for column N=ln
        #pragma unroll
        for (int fn = 0; fn < 4; ++fn) {
            const bf16* p = b0 + (wn * 64 + fn * 16 + ln) * LDT + 8 * g;
            bq[fn].h[0] = *(const v8bf*)(p);
            bq[fn].h[1] = *(const v8bf*)(p + 16);
        }
        #pragma unroll
        for (int fm = 0; fm < 2; ++fm)
            #pragma unroll
            for (int fn = 0; fn < 4; ++fn)
                acc[fm][fn] = __builtin_amdgcn_wmma_f32_16x16x32_bf16(
                    false, a[fm].v, false, bq[fn].v, (short)0, acc[fm][fn], false, false);

        waitTiles();          // next tile's DMA has landed in LDS
        __syncthreads();
    }

    // ---- fused LoRA rank-16 correction: one extra zero-padded WMMA per tile ----
    {
        AF a[2], bq[4];
        #pragma unroll
        for (int fm = 0; fm < 2; ++fm) {
            const int m = mBlk + wm * 32 + fm * 16 + ln;
            a[fm].h[0] = *(const v8bf*)(pb + (size_t)m * R_ + 8 * g);  // K<16 = proj
            #pragma unroll
            for (int i = 8; i < 16; ++i) a[fm].v[i] = (bf16)0.0f;      // K>=16 = 0
        }
        #pragma unroll
        for (int fn = 0; fn < 4; ++fn) {
            const int n = nBlk + wn * 64 + fn * 16 + ln;
            bq[fn].h[0] = *(const v8bf*)(lb + (size_t)n * R_ + 8 * g);
            #pragma unroll
            for (int i = 8; i < 16; ++i)                                // K>=16 -> 0
                bq[fn].v[i] = (bf16)0.0f;
        }
        #pragma unroll
        for (int fm = 0; fm < 2; ++fm)
            #pragma unroll
            for (int fn = 0; fn < 4; ++fn)
                acc[fm][fn] = __builtin_amdgcn_wmma_f32_16x16x32_bf16(
                    false, a[fm].v, false, bq[fn].v, (short)0, acc[fm][fn], false, false);
    }

    // ---- bias + store ----
    #pragma unroll
    for (int fn = 0; fn < 4; ++fn) {
        const int n  = nBlk + wn * 64 + fn * 16 + ln;
        const float bv = bias[n];
        #pragma unroll
        for (int fm = 0; fm < 2; ++fm) {
            const int m0 = mBlk + wm * 32 + fm * 16 + 8 * g;   // C layout: M = j + 8*g
            #pragma unroll
            for (int j = 0; j < 8; ++j)
                out[(size_t)(m0 + j) * DOUT + n] = acc[fm][fn][j] + bv;
        }
    }
}

// ------------------------------------------------------------------
extern "C" void kernel_launch(void* const* d_in, const int* in_sizes, int n_in,
                              void* d_out, int out_size, void* d_ws, size_t ws_size,
                              hipStream_t stream) {
    const float* x   = (const float*)d_in[0];
    const float* W   = (const float*)d_in[1];
    const float* bv  = (const float*)d_in[2];
    const float* Ash = (const float*)d_in[3];
    const float* Aex = (const float*)d_in[4];
    const float* lB  = (const float*)d_in[5];
    const float* gw1 = (const float*)d_in[6];
    const float* gb1 = (const float*)d_in[7];
    const float* gw2 = (const float*)d_in[8];
    const float* gb2 = (const float*)d_in[9];
    float* out = (float*)d_out;

    char* ws = (char*)d_ws;
    bf16*  xb     = (bf16*)(ws + WS_XB);
    bf16*  wb     = (bf16*)(ws + WS_WB);
    bf16*  lb     = (bf16*)(ws + WS_LB);
    bf16*  pb     = (bf16*)(ws + WS_PB);
    float* acomb  = (float*)(ws + WS_ACOMB);
    float* pooled = (float*)(ws + WS_POOLED);
    float* wts    = (float*)(ws + WS_WTS);

    zero_pooled_kernel<<<(B_ * DIN) / 256, 256, 0, stream>>>(pooled);
    convert_bf16_kernel<<<(size_t)DOUT * DIN / (256 * 8), 256, 0, stream>>>(W, wb);
    convert_bf16_kernel<<<(size_t)DOUT * R_ / (256 * 8), 256, 0, stream>>>(lB, lb);
    convert_x_pool_kernel<<<dim3(DIN / 256, S_ / 128, B_), 256, 0, stream>>>(x, xb, pooled);
    gate_kernel<<<1, 256, 0, stream>>>(pooled, gw1, gb1, gw2, gb2, wts);
    acomb_kernel<<<(B_ * R_ * DIN) / 256, 256, 0, stream>>>(Ash, Aex, wts, acomb);
    proj_kernel<<<M_TOT, 256, 0, stream>>>(xb, acomb, pb);
    gemm_kernel<<<dim3(DOUT / BN, M_TOT / BM), 256, 0, stream>>>(xb, wb, bv, pb, lb, out);
}
